// QuantumEnhancedQLSTM_65481071396915
// MI455X (gfx1250) — matrix operand order repaired and checked by reading, not compiled, not run on hardware
//
#include <hip/hip_runtime.h>

typedef __attribute__((ext_vector_type(16))) _Float16 v16h;
typedef __attribute__((ext_vector_type(8)))  _Float16 v8h;
typedef __attribute__((ext_vector_type(8)))  float    v8f;

#define NT     512          // timesteps
#define NB     64           // batch
#define DH     1024         // hidden
#define KD     2048         // D_IN + DH (reduction dim)
#define GN     4096         // 4*DH gate columns
#define ELEMS  (NB * DH)    // 65536
#define NBLK   32
#define NTHR   256
#define TOTTHR (NBLK * NTHR)  // 8192
#define LPAD   40             // 32 + 8 pad (f16 units) -> 80B row stride, 16B aligned

__device__ __forceinline__ float sigf(float v)  { return 1.0f / (1.0f + __expf(-v)); }
__device__ __forceinline__ float tanhv(float v) { return 1.0f - 2.0f / (1.0f + __expf(2.0f * v)); }

__device__ __forceinline__ void splitf(float v, _Float16& hi, _Float16& lo) {
  hi = (_Float16)v;
  lo = (_Float16)(v - (float)hi);
}

// Async copy of 16B global -> LDS (ASYNCcnt-tracked, bypasses VGPRs).
// LDS byte offset = low 32 bits of the generic pointer (ISA: LDS aperture
// addresses truncate to addr[31:0]).
__device__ __forceinline__ void async_fill16(const _Float16* g, _Float16* l) {
  unsigned loff = (unsigned)(uintptr_t)(const void*)l;
  asm volatile("global_load_async_to_lds_b128 %0, %1, off"
               :: "v"(loff), "v"(g)
               : "memory");
}
__device__ __forceinline__ void wait_async0() {
  asm volatile("s_wait_asynccnt 0x0" ::: "memory");
}

// Device-wide sense/generation barrier for the persistent kernel (32 co-resident blocks).
__device__ __forceinline__ void gridbar(unsigned* cnt, unsigned* gen) {
  __syncthreads();
  if (threadIdx.x == 0) {
    __threadfence();
    unsigned g = __hip_atomic_load(gen, __ATOMIC_RELAXED, __HIP_MEMORY_SCOPE_AGENT);
    if (atomicAdd(cnt, 1u) == (unsigned)(NBLK - 1)) {
      __hip_atomic_store(cnt, 0u, __ATOMIC_RELAXED, __HIP_MEMORY_SCOPE_AGENT);
      __hip_atomic_store(gen, g + 1u, __ATOMIC_RELEASE, __HIP_MEMORY_SCOPE_AGENT);
    } else {
      while (__hip_atomic_load(gen, __ATOMIC_ACQUIRE, __HIP_MEMORY_SCOPE_AGENT) == g)
        __builtin_amdgcn_s_sleep(2);
    }
    __threadfence();
  }
  __syncthreads();
}

__global__ void bar_init(unsigned* bar) { bar[0] = 0u; bar[1] = 0u; }

// Split fp32 W into hi/lo f16 planes (Dekker decomposition), row-major [GN][KD].
__global__ __launch_bounds__(NTHR) void w_split(const float* __restrict__ W,
                                                _Float16* __restrict__ Wh,
                                                _Float16* __restrict__ Wl) {
  int i = (int)(blockIdx.x * NTHR + threadIdx.x) * 4;
  #pragma unroll
  for (int j = 0; j < 4; ++j) {
    _Float16 h, l;
    splitf(W[i + j], h, l);
    Wh[i + j] = h;
    Wl[i + j] = l;
  }
}

__global__ __launch_bounds__(NTHR) void lstm_persist(
    const float* __restrict__ x, const float* __restrict__ hx0,
    const float* __restrict__ cx0, const float* __restrict__ bias,
    const _Float16* __restrict__ Wh, const _Float16* __restrict__ Wl,
    _Float16* __restrict__ xhh, _Float16* __restrict__ xhl,
    float* __restrict__ cells, float* __restrict__ gates,
    float* __restrict__ out, unsigned* __restrict__ bar) {
  // Double-buffered split-f16 slab of combined[0:64, k0:k0+32] (20.5 KB of 320 KB).
  __shared__ _Float16 lAh[2][NB][LPAD];
  __shared__ _Float16 lAl[2][NB][LPAD];

  const int tid    = (int)threadIdx.x;
  const int lane   = tid & 31;
  const int wid    = (int)blockIdx.x * 8 + (tid >> 5);   // 0..255 wave id
  const int n0     = wid << 4;                           // 16-wide gate-column strip
  const int ncol   = n0 + (lane & 15);
  const int koff8  = (lane & 16) ? 8 : 0;                // unused; A frag uses koff16 too
  const int koff16 = (lane & 16) ? 16 : 0;               // B-frag K half select
  const int rbase  = (lane & 16) ? 8 : 0;                // C/D M offset for hi lanes
  const int gtid   = (int)blockIdx.x * NTHR + tid;
  const int fr     = tid >> 2;                           // LDS fill row 0..63
  const int fq     = (tid & 3) * 8;                      // LDS fill col chunk

  // ---- Prologue: stage combined[x_0 | h_0] as split-f16, copy c_0 ----
  {
    int e = gtid;
    #pragma unroll
    for (int i = 0; i < 8; ++i, e += TOTTHR) {
      int n = e >> 10, h = e & 1023;
      _Float16 hh, ll;
      splitf(x[e], hh, ll);
      xhh[n * KD + h] = hh;           xhl[n * KD + h] = ll;
      splitf(hx0[e], hh, ll);
      xhh[n * KD + DH + h] = hh;      xhl[n * KD + DH + h] = ll;
      cells[e] = cx0[e];
    }
  }
  gridbar(bar, bar + 1);

  const _Float16* wrh = Wh + (size_t)ncol * KD;
  const _Float16* wrl = Wl + (size_t)ncol * KD;
  const _Float16* srcH = xhh + fr * KD + fq;
  const _Float16* srcL = xhl + fr * KD + fq;

  for (int t = 0; t < NT; ++t) {
    // ======== GEMM phase: gates = combined @ W^T + b (split-f16 WMMA) ========
    const float bv = bias[ncol];            // all 8 C elements share column ncol
    v8f acc[4];
    #pragma unroll
    for (int r = 0; r < 8; ++r) { acc[0][r] = bv; acc[1][r] = bv; acc[2][r] = bv; acc[3][r] = bv; }

    // Kick off async fill of buffer 0 for k0 = 0.
    async_fill16(srcH, &lAh[0][fr][fq]);
    async_fill16(srcL, &lAl[0][fr][fq]);

    for (int kc = 0; kc < KD / 32; ++kc) {
      const int k0  = kc * 32;
      const int cur = kc & 1;

      wait_async0();        // my fills of buf[cur] have landed in LDS
      __syncthreads();      // everyone's fills landed; everyone done reading buf[!cur]

      if (kc + 1 < KD / 32) {       // overlap next chunk's DMA with the WMMAs
        async_fill16(srcH + (k0 + 32), &lAh[cur ^ 1][fr][fq]);
        async_fill16(srcL + (k0 + 32), &lAl[cur ^ 1][fr][fq]);
      }

      // B fragments: 16 contiguous f16 from W row `ncol`, K half by lane group
      v16h bh = *(const v16h*)(wrh + k0 + koff16);
      v16h bl = *(const v16h*)(wrl + k0 + koff16);

      int kp = (k0 + 32) & (KD - 1);        // prefetch next K chunk of W
      __builtin_prefetch(wrh + kp, 0, 0);
      __builtin_prefetch(wrl + kp, 0, 0);

      #pragma unroll
      for (int mt = 0; mt < 4; ++mt) {
        const _Float16* ah = &lAh[cur][mt * 16 + (lane & 15)][koff16];
        const _Float16* al = &lAl[cur][mt * 16 + (lane & 15)][koff16];
        v8h ah0 = *(const v8h*)ah, ah1 = *(const v8h*)(ah + 8);
        v8h al0 = *(const v8h*)al, al1 = *(const v8h*)(al + 8);
        v16h afh = __builtin_shufflevector(ah0, ah1, 0,1,2,3,4,5,6,7,8,9,10,11,12,13,14,15);
        v16h afl = __builtin_shufflevector(al0, al1, 0,1,2,3,4,5,6,7,8,9,10,11,12,13,14,15);
        // Dekker product: Ah*Bh + Ah*Bl + Al*Bh, f32 accumulate (~fp32 accuracy)
        acc[mt] = __builtin_amdgcn_wmma_f32_16x16x32_f16(false, afh, false, bh, (short)0, acc[mt], false, false);
        acc[mt] = __builtin_amdgcn_wmma_f32_16x16x32_f16(false, afh, false, bl, (short)0, acc[mt], false, false);
        acc[mt] = __builtin_amdgcn_wmma_f32_16x16x32_f16(false, afl, false, bh, (short)0, acc[mt], false, false);
      }
    }

    // Scatter D tiles to the gate buffer [NB][GN]
    #pragma unroll
    for (int mt = 0; mt < 4; ++mt) {
      float* gp = gates + (size_t)(mt * 16 + rbase) * GN + ncol;
      #pragma unroll
      for (int r = 0; r < 8; ++r) gp[(size_t)r * GN] = acc[mt][r];
    }
    gridbar(bar, bar + 1);

    // ======== Elementwise LSTM update + staging of next combined ========
    {
      int e = gtid;
      const float* xnext = x + (size_t)(t + 1) * ELEMS;
      #pragma unroll
      for (int i = 0; i < 8; ++i, e += TOTTHR) {
        int n = e >> 10, h = e & 1023;
        const float* gr = gates + (size_t)n * GN;
        float fg = sigf(gr[h]);
        float ig = sigf(gr[DH + h]);
        float gg = tanhv(gr[2 * DH + h]);
        float og = sigf(gr[3 * DH + h]);
        float c  = fg * cells[e] + ig * gg;
        float hy = og * tanhv(c);
        cells[e] = c;
        out[(size_t)t * ELEMS + e] = hy;

        _Float16 hh, ll;
        splitf(hy, hh, ll);
        xhh[n * KD + DH + h] = hh;  xhl[n * KD + DH + h] = ll;
        if (t + 1 < NT) {
          splitf(xnext[e], hh, ll);
          xhh[n * KD + h] = hh;     xhl[n * KD + h] = ll;
        } else {
          out[(size_t)NT * ELEMS + e]         = hy;   // final hx
          out[(size_t)NT * ELEMS + ELEMS + e] = c;    // final cx
        }
      }
    }
    gridbar(bar, bar + 1);
  }
}

extern "C" void kernel_launch(void* const* d_in, const int* in_sizes, int n_in,
                              void* d_out, int out_size, void* d_ws, size_t ws_size,
                              hipStream_t stream) {
  const float* x   = (const float*)d_in[0];   // [T, B, D_IN]
  const float* hx0 = (const float*)d_in[1];   // [B, H]
  const float* cx0 = (const float*)d_in[2];   // [B, H]
  const float* W   = (const float*)d_in[3];   // [4H, D_IN+H]
  const float* b   = (const float*)d_in[4];   // [4H]
  float* out = (float*)d_out;

  char* p = (char*)d_ws;
  auto carve = [&](size_t bytes) { void* r = (void*)p; p += (bytes + 255) & ~(size_t)255; return r; };
  _Float16* Wh    = (_Float16*)carve((size_t)GN * KD * 2);   // 16 MB
  _Float16* Wl    = (_Float16*)carve((size_t)GN * KD * 2);   // 16 MB
  _Float16* xhh   = (_Float16*)carve((size_t)NB * KD * 2);   // 256 KB
  _Float16* xhl   = (_Float16*)carve((size_t)NB * KD * 2);   // 256 KB
  float*    cells = (float*)carve((size_t)ELEMS * 4);        // 256 KB
  float*    gates = (float*)carve((size_t)NB * GN * 4);      // 1 MB
  unsigned* bar   = (unsigned*)carve(256);

  bar_init<<<1, 1, 0, stream>>>(bar);
  w_split<<<(GN * KD) / (NTHR * 4), NTHR, 0, stream>>>(W, Wh, Wl);
  lstm_persist<<<NBLK, NTHR, 0, stream>>>(x, hx0, cx0, b, Wh, Wl, xhh, xhl,
                                          cells, gates, out, bar);
  (void)in_sizes; (void)n_in; (void)out_size; (void)ws_size;
}
